// BayesHead_35553739276746
// MI455X (gfx1250) — matrix operand order repaired and checked by reading, not compiled, not run on hardware
//
#include <hip/hip_runtime.h>

#define B_ 4
#define T_ 4096
#define C_ 1024
#define H_ 64

typedef __attribute__((ext_vector_type(16))) __bf16        v16bf;
typedef __attribute__((ext_vector_type(8)))  float         v8f;
typedef __attribute__((ext_vector_type(4)))  float         v4f;
typedef __attribute__((ext_vector_type(4)))  unsigned int  v4u;

// 16 bf16 = 8 VGPRs = one WMMA A/B operand.
union ABf {
    v16bf   v;
    __bf16  e[16];
    v4u     q[2];
};

__device__ __forceinline__ float redmax16(float x) {
    x = fmaxf(x, __shfl_xor(x, 1, 32));
    x = fmaxf(x, __shfl_xor(x, 2, 32));
    x = fmaxf(x, __shfl_xor(x, 4, 32));
    x = fmaxf(x, __shfl_xor(x, 8, 32));
    return x;
}
__device__ __forceinline__ float redsum16(float x) {
    x += __shfl_xor(x, 1, 32);
    x += __shfl_xor(x, 2, 32);
    x += __shfl_xor(x, 4, 32);
    x += __shfl_xor(x, 8, 32);
    return x;
}

__device__ __forceinline__ v8f wmma_bf16(v16bf a, v16bf b, v8f c) {
    return __builtin_amdgcn_wmma_f32_16x16x32_bf16(
        /*neg_a=*/false, a, /*neg_b=*/false, b,
        /*c_mod=*/(short)0, c, /*reuse_a=*/false, /*reuse_b=*/false);
}

// ---------------------------------------------------------------------------
// Kernel 0: one-shot fp32 -> bf16 conversion of Wq|Wk|Wv into Wb[3][H][C].
// 384 KB of bf16 weights; stays L2-resident for the projection pass.
// ---------------------------------------------------------------------------
__global__ __launch_bounds__(256)
void wconv_kernel(const float* __restrict__ Wq, const float* __restrict__ Wk,
                  const float* __restrict__ Wv, __bf16* __restrict__ Wb)
{
    const int i   = (blockIdx.x * 256 + threadIdx.x) * 4;   // element index
    const int per = H_ * C_;                                // 65536 per matrix
    const int mat = i / per;
    const int off = i - mat * per;
    const float* W = (mat == 0) ? Wq : (mat == 1) ? Wk : Wv;
    v4f x = *(const v4f*)(W + off);
    __bf16* d = Wb + i;
    d[0] = (__bf16)x[0];
    d[1] = (__bf16)x[1];
    d[2] = (__bf16)x[2];
    d[3] = (__bf16)x[3];
}

// ---------------------------------------------------------------------------
// Kernel 1: Q/K/V projection, fp32 activations -> bf16, WMMA 16x16x32.
// One wave per 16-row tile; blockIdx.y selects Q / K / V. K-loop outer with
// 4 accumulators: each A tile is converted once and reused by 4 WMMAs whose
// B operands are pure 16-byte loads of pre-converted weights.
// Q,K stored [B,T,H]; V stored transposed [B,H,T] (stride-based, branch-free).
// ---------------------------------------------------------------------------
__global__ __launch_bounds__(32)
void proj_kernel(const float* __restrict__ q, const float* __restrict__ k,
                 const float* __restrict__ v, const __bf16* __restrict__ Wb,
                 __bf16* __restrict__ Qb, __bf16* __restrict__ Kb,
                 __bf16* __restrict__ Vt)
{
    const int mat  = blockIdx.y;            // 0=Q, 1=K, 2=V
    const int tile = blockIdx.x;            // B*T/16 tiles
    const int b    = tile >> 8;             // 256 tiles per batch
    const int tb   = (tile & 255) << 4;
    const int lane = threadIdx.x;
    const int m    = lane & 15;
    const int half = lane >> 4;

    const float* src = (mat == 0) ? q : (mat == 1) ? k : v;
    const float* arow = src + ((size_t)b * T_ + tb + m) * C_;

    // B-layout base: lane n = m selects W row; lanes 16-31 hold K=8..15,24..31.
    const __bf16* wbase = Wb + (size_t)mat * H_ * C_ + (size_t)m * C_ + 8 * half;

    v8f acc[4] = {};
    for (int kc = 0; kc < C_ / 32; ++kc) {
        const int c0 = kc * 32;
        // ---- A (16x32 bf16 A-layout): convert once, reuse for 4 N-tiles ----
        v4f a0 = *(const v4f*)(arow + c0 + 8 * half);
        v4f a1 = *(const v4f*)(arow + c0 + 8 * half + 4);
        v4f a2 = *(const v4f*)(arow + c0 + 16 + 8 * half);
        v4f a3 = *(const v4f*)(arow + c0 + 16 + 8 * half + 4);
        ABf A;
        #pragma unroll
        for (int i = 0; i < 4; ++i) {
            A.e[i]      = (__bf16)a0[i];
            A.e[4 + i]  = (__bf16)a1[i];
            A.e[8 + i]  = (__bf16)a2[i];
            A.e[12 + i] = (__bf16)a3[i];
        }
        // ---- 4 N-tiles over H=64: pure loads of pre-converted weights ----
        #pragma unroll
        for (int nt = 0; nt < 4; ++nt) {
            const __bf16* wp = wbase + (size_t)(nt * 16) * C_ + c0;
            ABf Bt;
            Bt.q[0] = *(const v4u*)(wp);
            Bt.q[1] = *(const v4u*)(wp + 16);
            acc[nt] = wmma_bf16(A.v, Bt.v, acc[nt]);
        }
    }

    // C/D layout: lane holds column n=m for rows r + 8*half. Branch-free
    // stride-selected store: V goes to [B,H,T], Q/K to [B,T,H].
    __bf16* dst = ((mat == 0) ? Qb : (mat == 1) ? Kb : Vt) + (size_t)b * T_ * H_;
    const int rs = (mat == 2) ? 1 : H_;     // stride of the T index
    const int cs = (mat == 2) ? T_ : 1;     // stride of the H index
    #pragma unroll
    for (int nt = 0; nt < 4; ++nt) {
        const int h = nt * 16 + m;
        #pragma unroll
        for (int r = 0; r < 8; ++r) {
            const int trow = tb + r + 8 * half;
            dst[(size_t)trow * rs + (size_t)h * cs] = (__bf16)acc[nt][r];
        }
    }
}

// ---------------------------------------------------------------------------
// Kernel 2: flash attention, causal mask tril(diag=1), effective scale 1/64.
// One wave per 16-query tile; 32-key blocks; fp32 online softmax computed in
// the exp2 domain; P transposed C-layout -> A-layout through 1 KB of LDS.
// ---------------------------------------------------------------------------
__global__ __launch_bounds__(32)
void attn_kernel(const __bf16* __restrict__ Qb, const __bf16* __restrict__ Kb,
                 const __bf16* __restrict__ Vt, float* __restrict__ out)
{
    __shared__ __bf16 Pld[16 * 32];
    const int qtile = blockIdx.x;
    const int b     = blockIdx.y;
    const int qbase = qtile * 16;
    const int lane  = threadIdx.x;
    const int m     = lane & 15;
    const int half  = lane >> 4;
    // exp2-domain scale: (H^-0.5 applied twice) * log2(e)
    const float SCL = (1.0f / 64.0f) * 1.4426950408889634f;

    // Q tile (16 x 64) in A-layout, two K=32 chunks over h.
    ABf QA[2];
    {
        const __bf16* qrow = Qb + ((size_t)b * T_ + qbase + m) * H_;
        #pragma unroll
        for (int hk = 0; hk < 2; ++hk) {
            QA[hk].q[0] = *(const v4u*)(qrow + hk * 32 + 8 * half);
            QA[hk].q[1] = *(const v4u*)(qrow + hk * 32 + 16 + 8 * half);
        }
    }

    float mrow[8], lsum[8];
    v8f O[4] = {};
    #pragma unroll
    for (int r = 0; r < 8; ++r) { mrow[r] = -3.0e38f; lsum[r] = 0.0f; }

    int smax = qbase + 16;                  // furthest visible key (s <= t+1)
    if (smax > T_ - 1) smax = T_ - 1;
    const int jmax = smax >> 5;

    for (int jb = 0; jb <= jmax; ++jb) {
        const int kbase = jb << 5;

        // ---- S = Q K^T : two 16x16 N-tiles, K=64 split into 2 WMMAs ----
        v8f S0 = {}, S1 = {};
        #pragma unroll
        for (int hk = 0; hk < 2; ++hk) {
            ABf BK0, BK1;
            const __bf16* k0 =
                Kb + ((size_t)b * T_ + kbase + m) * H_ + hk * 32 + 8 * half;
            const __bf16* k1 = k0 + (size_t)16 * H_;
            BK0.q[0] = *(const v4u*)(k0);
            BK0.q[1] = *(const v4u*)(k0 + 16);
            BK1.q[0] = *(const v4u*)(k1);
            BK1.q[1] = *(const v4u*)(k1 + 16);
            S0 = wmma_bf16(QA[hk].v, BK0.v, S0);
            S1 = wmma_bf16(QA[hk].v, BK1.v, S1);
        }

        if (jb < jmax)  // warm the path for the next K block
            __builtin_prefetch(Kb + ((size_t)b * T_ + kbase + 32 + m) * H_, 0, 0);

        // ---- masked online softmax, exp2 domain (branch-free) ----
        float p0[8], p1[8], mnew[8];
        #pragma unroll
        for (int r = 0; r < 8; ++r) {
            const int t  = qbase + r + 8 * half;
            const int s0 = kbase + m;
            const int s1 = kbase + 16 + m;
            float x0 = S0[r] * SCL;
            float x1 = S1[r] * SCL;
            if (s0 > t + 1) x0 = -3.0e38f;
            if (s1 > t + 1) x1 = -3.0e38f;
            const float rm = redmax16(fmaxf(x0, x1));
            mnew[r] = fmaxf(mrow[r], rm);
            p0[r] = exp2f(x0 - mnew[r]);
            p1[r] = exp2f(x1 - mnew[r]);
        }
        #pragma unroll
        for (int r = 0; r < 8; ++r) {
            const float alpha = exp2f(mrow[r] - mnew[r]);
            const float rs    = redsum16(p0[r] + p1[r]);
            lsum[r] = lsum[r] * alpha + rs;
            mrow[r] = mnew[r];
            #pragma unroll
            for (int ht = 0; ht < 4; ++ht) O[ht][r] *= alpha;
        }

        // ---- transpose P (C-layout -> A-layout) through LDS ----
        #pragma unroll
        for (int r = 0; r < 8; ++r) {
            Pld[(r + 8 * half) * 32 + m]      = (__bf16)p0[r];
            Pld[(r + 8 * half) * 32 + 16 + m] = (__bf16)p1[r];
        }
        __syncthreads();
        ABf PA;
        PA.q[0] = *(const v4u*)(&Pld[m * 32 + 8 * half]);
        PA.q[1] = *(const v4u*)(&Pld[m * 32 + 16 + 8 * half]);
        __syncthreads();

        // ---- O += P V : 4 N-tiles over H; V B-tiles contiguous via Vt ----
        #pragma unroll
        for (int ht = 0; ht < 4; ++ht) {
            ABf BV;
            const __bf16* vp =
                Vt + ((size_t)b * H_ + ht * 16 + m) * T_ + kbase + 8 * half;
            BV.q[0] = *(const v4u*)(vp);
            BV.q[1] = *(const v4u*)(vp + 16);
            O[ht] = wmma_bf16(PA.v, BV.v, O[ht]);
        }
    }

    // ---- normalize and store fp32 output [B,T,H] ----
    #pragma unroll
    for (int r = 0; r < 8; ++r) {
        const int t = qbase + r + 8 * half;
        const float inv = 1.0f / lsum[r];
        #pragma unroll
        for (int ht = 0; ht < 4; ++ht)
            out[((size_t)b * T_ + t) * H_ + ht * 16 + m] = O[ht][r] * inv;
    }
}

extern "C" void kernel_launch(void* const* d_in, const int* in_sizes, int n_in,
                              void* d_out, int out_size, void* d_ws, size_t ws_size,
                              hipStream_t stream) {
    const float* q  = (const float*)d_in[0];
    const float* k  = (const float*)d_in[1];
    const float* v  = (const float*)d_in[2];
    const float* Wq = (const float*)d_in[3];
    const float* Wk = (const float*)d_in[4];
    const float* Wv = (const float*)d_in[5];
    float* out = (float*)d_out;

    const size_t elems = (size_t)B_ * T_ * H_;   // 1M elements per matrix
    __bf16* Qb = (__bf16*)d_ws;                  // bf16 [B,T,H]
    __bf16* Kb = Qb + elems;                     // bf16 [B,T,H]
    __bf16* Vt = Kb + elems;                     // bf16 [B,H,T] (transposed)
    __bf16* Wb = Vt + elems;                     // bf16 [3,H,C]

    wconv_kernel<<<dim3(3 * H_ * C_ / (256 * 4)), 256, 0, stream>>>(Wq, Wk, Wv, Wb);

    dim3 pg(B_ * T_ / 16, 3);
    proj_kernel<<<pg, 32, 0, stream>>>(q, k, v, Wb, Qb, Kb, Vt);

    dim3 ag(T_ / 16, B_);
    attn_kernel<<<ag, 32, 0, stream>>>(Qb, Kb, Vt, out);
}
